// Moe_70626442215504
// MI455X (gfx1250) — matrix operand order repaired and checked by reading, not compiled, not run on hardware
//
#include <hip/hip_runtime.h>
#include <hip/hip_bf16.h>

// ---------------------------------------------------------------------------
// MoE (DeepSeek-style group-limited routing) for MI455X / gfx1250.
// T=2048 D=1024 I=512 E=32 G=8 LG=4 K=6, shared expert SI=1024.
// bf16 WMMA (v_wmma_f32_16x16x32_bf16) path, fp32 accumulation,
// double-buffered LDS weight slabs, native packed bf16 converts,
// compile-time weight strides for immediate-offset staging loads.
// ---------------------------------------------------------------------------

#define T_TOK 2048
#define DDIM  1024
#define IDIM  512
#define NEXP  32
#define NGRP  8
#define KTOP  6
#define TM    64          // M tile (rows) per GEMM block

typedef __attribute__((ext_vector_type(16))) __bf16 v16bf;
typedef __attribute__((ext_vector_type(8)))  float  v8f;

__device__ __forceinline__ __bf16 f2bf(float f) {
    return (__bf16)f;     // native v_cvt (RNE) on gfx1250
}

// -------------------------- init: y = sb3, cnt = 0 --------------------------
__global__ void moe_init(float* __restrict__ y, const float* __restrict__ sb3,
                         int* __restrict__ cnt) {
    long i = (long)blockIdx.x * 256 + threadIdx.x;
    if (i < (long)T_TOK * DDIM) y[i] = sb3[i & (DDIM - 1)];
    if (i < NEXP) cnt[i] = 0;
}

// -------------------------- router: 1 wave per token ------------------------
__global__ void moe_router(const float* __restrict__ x,
                           const float* __restrict__ rw,
                           const float* __restrict__ rb,
                           int* __restrict__ cnt,
                           int* __restrict__ tlist,
                           float* __restrict__ glist) {
    int t = blockIdx.x;
    int lane = threadIdx.x;                       // 32 lanes, lane == expert
    const float4* x4 = (const float4*)(x + (long)t * DDIM);
    const float4* w4 = (const float4*)(rw + (long)lane * DDIM);
    float acc = 0.f;
    #pragma unroll 4
    for (int i = 0; i < DDIM / 4; ++i) {
        float4 a = x4[i], b = w4[i];
        acc += a.x * b.x + a.y * b.y + a.z * b.z + a.w * b.w;
    }
    acc += rb[lane];
    // softmax across the wave (wave32)
    float mx = acc;
    for (int o = 16; o; o >>= 1) mx = fmaxf(mx, __shfl_xor(mx, o, 32));
    float ex = __expf(acc - mx);
    float sm = ex;
    for (int o = 16; o; o >>= 1) sm += __shfl_xor(sm, o, 32);
    float score = ex / sm;

    __shared__ float s[NEXP];
    s[lane] = score;
    __syncthreads();

    if (lane == 0) {
        // group score = sum of top-2 of each group of 4
        float gs[NGRP];
        #pragma unroll
        for (int g = 0; g < NGRP; ++g) {
            float m1 = -1e30f, m2 = -1e30f;
            #pragma unroll
            for (int j = 0; j < 4; ++j) {
                float v = s[g * 4 + j];
                if (v > m1) { m2 = m1; m1 = v; } else if (v > m2) m2 = v;
            }
            gs[g] = m1 + m2;
        }
        bool keep[NGRP] = {false,false,false,false,false,false,false,false};
        for (int it = 0; it < 4; ++it) {                  // top-4 groups
            float best = -1e30f; int bi = 0;
            for (int g = 0; g < NGRP; ++g)
                if (!keep[g] && gs[g] > best) { best = gs[g]; bi = g; }
            keep[bi] = true;
        }
        float ms[NEXP];
        for (int e = 0; e < NEXP; ++e) ms[e] = keep[e >> 2] ? s[e] : -1e30f;
        for (int k = 0; k < KTOP; ++k) {                  // top-6 experts
            float best = -1e30f; int bi = 0;
            for (int e = 0; e < NEXP; ++e)
                if (ms[e] > best) { best = ms[e]; bi = e; }
            ms[bi] = -1e30f;
            float gate = s[bi];                           // ROUTE_SCALE = 1
            int pos = atomicAdd(&cnt[bi], 1);
            tlist[(long)bi * T_TOK + pos] = t;
            glist[(long)bi * T_TOK + pos] = gate;
        }
    }
}

// -------------------------- scan: pad + offsets -----------------------------
__global__ void moe_scan(const int* __restrict__ cnt,
                         int* __restrict__ paddedCnt,
                         int* __restrict__ rowStartTile) {
    if (threadIdx.x == 0) {
        int tot = 0;
        for (int e = 0; e < NEXP; ++e) {
            int p = (cnt[e] + (TM - 1)) & ~(TM - 1);
            paddedCnt[e] = p;
            rowStartTile[e] = tot >> 4;                  // in 16-row tiles
            tot += p;
        }
    }
}

// ---- pack A: gather token rows -> bf16 in WMMA 16-bit A fragment layout ----
// A layout (16x16x32 bf16, per ISA table): lane<16 row M=lane, halves 0..7 =
// K 0..7, halves 8..15 = K 16..23; lane>=16 row M=lane-16, K 8..15 / 24..31.
__global__ __launch_bounds__(256)
void moe_pack_a(const float* __restrict__ x,
                __bf16* __restrict__ Abuf,
                const int* __restrict__ rowStartTile,   // null -> 0
                const int* __restrict__ cnts,           // null -> fixedCnt
                const int* __restrict__ paddedCnt,      // null -> fixedCnt
                int fixedCnt,
                int* __restrict__ tlist,                // null for shared
                float* __restrict__ glist) {
    int e = blockIdx.x, mt = blockIdx.y;
    int padded = paddedCnt ? paddedCnt[e] : fixedCnt;
    if (mt * TM >= padded) return;
    int cnt = cnts ? cnts[e] : fixedCnt;
    long tileBase = (rowStartTile ? rowStartTile[e] : 0) + mt * 4;
    int tid = threadIdx.x;
    for (int rep = 0; rep < 16; ++rep) {
        int fl  = tid + rep * 256;                      // 0..4095 frag-lanes
        int t16 = fl >> 10;                             // 0..3
        int kt  = (fl >> 5) & 31;                       // 0..31
        int l   = fl & 31;
        int rloc = (t16 << 4) + (l & 15);
        int r = mt * TM + rloc;
        int tok = 0;
        bool vld = true;
        if (tlist) { if (r < cnt) tok = tlist[(long)e * T_TOK + r]; else vld = false; }
        else tok = r;
        const float* xr = x + (long)tok * DDIM + kt * 32 + ((l >> 4) & 1) * 8;
        __bf16* dst = &Abuf[((tileBase + t16) * 32 + kt) * 512 + (long)l * 16];
        #pragma unroll
        for (int h = 0; h < 16; ++h) {
            int kk = (h & 7) + 16 * (h >> 3);           // immediate offsets
            dst[h] = vld ? f2bf(xr[kk]) : (__bf16)0.f;
        }
        if (tlist && kt == 0 && l < 16 && r >= cnt) {   // zero the padded slots
            tlist[(long)e * T_TOK + r] = 0;
            glist[(long)e * T_TOK + r] = 0.f;
        }
    }
}

// -------- fused expert: h=silu((A W1+b1)*(A W2+b2)); y += gate*(h W3+b3) ----
// LD12 = row stride of W1/W2 (compile-time so staging loads use imm offsets).
template <int LD12>
__global__ __launch_bounds__(256)
void moe_expert_gemm(const __bf16* __restrict__ Abuf,
                     const int* __restrict__ rowStartTile,   // null -> 0
                     const int* __restrict__ paddedCnt,      // null -> fixedCnt
                     int fixedCnt,
                     const int* __restrict__ tlist,          // null -> identity
                     const float* __restrict__ glist,        // null -> 1.0
                     const float* __restrict__ W1, long sW1,
                     const float* __restrict__ W2, long sW2,
                     const float* __restrict__ b1, long sB12,
                     const float* __restrict__ b2,
                     const float* __restrict__ W3, long sW3,
                     const float* __restrict__ b3, long sB3, // b3 null -> 0
                     float* __restrict__ y) {
    const int e = blockIdx.x, mt = blockIdx.y;
    const int cnt = paddedCnt ? paddedCnt[e] : fixedCnt;
    if (mt * TM >= cnt) return;

    const int tid = threadIdx.x, lane = tid & 31, w = tid >> 5;
    const int m16 = w & 3;            // wave's 16-row subtile (0..3)
    const int nh  = w >> 2;           // wave's N half (0..1)
    const long gTileBase = (rowStartTile ? rowStartTile[e] : 0) + mt * 4;

    __shared__ alignas(32) __bf16 Hs[4 * 16 * 32 * 16];   // 64 KB: h in A-swizzle
    __shared__ alignas(32) __bf16 Bs[2][8192];            // 2 x 16 KB slab (dbl buf)
    __shared__ float gateS[TM];
    __shared__ int   tokS[TM];

    if (tid < TM) {
        int r = mt * TM + tid;
        if (tlist) { tokS[tid] = tlist[(long)e * T_TOK + r];
                     gateS[tid] = glist[(long)e * T_TOK + r]; }
        else       { tokS[tid] = r; gateS[tid] = 1.0f; }
    }

    const float* W1e = W1 + (long)e * sW1;
    const float* W2e = W2 + (long)e * sW2;
    const float* b1e = b1 + (long)e * sB12;
    const float* b2e = b2 + (long)e * sB12;
    const float* W3e = W3 + (long)e * sW3;
    const float* b3e = b3 ? (b3 + (long)e * sB3) : nullptr;

    // B-fragment staging layout: lane<16: K=0..7,16..23, lane>=16: K=8..15,24..31, N=lane&15
    auto stage12 = [&](int nc, int kt, __bf16* dst) {
        int nt = tid >> 5, pl = tid & 31;
        int n = nc * 128 + nt * 16 + (pl & 15);
        int kb = kt * 32 + ((pl >> 4) & 1) * 8;
        const float* p1 = W1e + (long)kb * LD12 + n;     // one addr setup,
        const float* p2 = W2e + (long)kb * LD12 + n;     // then imm offsets
        __bf16* d1 = dst + ((0 * 8 + nt) * 32 + pl) * 16;
        __bf16* d2 = dst + ((1 * 8 + nt) * 32 + pl) * 16;
        #pragma unroll
        for (int h = 0; h < 16; ++h) {
            d1[h] = f2bf(p1[((h & 7) + 16 * (h >> 3)) * LD12]);
            d2[h] = f2bf(p2[((h & 7) + 16 * (h >> 3)) * LD12]);
        }
    };
    auto stage3 = [&](int nc2, int kt, __bf16* dst) {
        #pragma unroll
        for (int rep = 0; rep < 2; ++rep) {              // slab 32x256
            int fl = tid + rep * 256;                    // 0..511 frag-lanes
            int nt = fl >> 5, pl = fl & 31;
            int n = nc2 * 256 + nt * 16 + (pl & 15);
            int kb = kt * 32 + ((pl >> 4) & 1) * 8;
            const float* p3 = W3e + (long)kb * DDIM + n;
            __bf16* d3 = dst + (nt * 32 + pl) * 16;
            #pragma unroll
            for (int h = 0; h < 16; ++h)
                d3[h] = f2bf(p3[((h & 7) + 16 * (h >> 3)) * DDIM]);
        }
    };

    // ---------------- stage 1: h1/h2 GEMMs over D=1024, N chunks of 128 -----
    for (int nc = 0; nc < 4; ++nc) {
        v8f c1[4], c2[4];
        #pragma unroll
        for (int j = 0; j < 4; ++j)
            for (int q = 0; q < 8; ++q) { c1[j][q] = 0.f; c2[j][q] = 0.f; }

        stage12(nc, 0, &Bs[0][0]);
        __syncthreads();
        for (int kt = 0; kt < 32; ++kt) {
            int p = kt & 1;
            if (kt + 1 < 32) stage12(nc, kt + 1, &Bs[p ^ 1][0]);  // prefetch next slab
            const v16bf a = *(const v16bf*)&Abuf[(((gTileBase + m16) * 32 + kt) * 32 + lane) * 16];
            #pragma unroll
            for (int j = 0; j < 4; ++j) {
                int nt = nh * 4 + j;
                v16bf bA = *(const v16bf*)&Bs[p][((0 * 8 + nt) * 32 + lane) * 16];
                v16bf bB = *(const v16bf*)&Bs[p][((1 * 8 + nt) * 32 + lane) * 16];
                c1[j] = __builtin_amdgcn_wmma_f32_16x16x32_bf16(false, a, false, bA,
                                                                (short)0, c1[j], false, false);
                c2[j] = __builtin_amdgcn_wmma_f32_16x16x32_bf16(false, a, false, bB,
                                                                (short)0, c2[j], false, false);
            }
            __syncthreads();
        }
        // epilogue: silu(h1*h2) -> Hs in A-swizzle (K dim = IDIM)
        #pragma unroll
        for (int j = 0; j < 4; ++j) {
            int nIC = nc * 128 + (nh * 4 + j) * 16 + (lane & 15);
            float bb1 = b1e[nIC], bb2 = b2e[nIC];
            int kt2 = nIC >> 5, kk2 = nIC & 31;
            int lhi = (kk2 >> 3) & 1;
            int half2 = (kk2 & 7) + 8 * (kk2 >> 4);
            #pragma unroll
            for (int r = 0; r < 8; ++r) {
                float h1v = c1[j][r] + bb1;
                float h2v = c2[j][r] + bb2;
                float v = h1v * h2v;
                float hv = v / (1.f + __expf(-v));       // silu
                int row = r + ((lane >> 4) << 3);        // 0..15 within m16
                int lane2 = row + (lhi << 4);
                Hs[((m16 * 16 + kt2) * 32 + lane2) * 16 + half2] = f2bf(hv);
            }
        }
    }

    // ---------------- stage 2: y += gate * (h @ W3 + b3) --------------------
    for (int nc2 = 0; nc2 < 4; ++nc2) {
        v8f co[8];
        #pragma unroll
        for (int j = 0; j < 8; ++j)
            for (int q = 0; q < 8; ++q) co[j][q] = 0.f;

        stage3(nc2, 0, &Bs[0][0]);
        __syncthreads();                                 // also orders Hs writes
        for (int kt = 0; kt < 16; ++kt) {
            int p = kt & 1;
            if (kt + 1 < 16) stage3(nc2, kt + 1, &Bs[p ^ 1][0]);
            const v16bf a = *(const v16bf*)&Hs[((m16 * 16 + kt) * 32 + lane) * 16];
            #pragma unroll
            for (int j = 0; j < 8; ++j) {
                int nt = nh * 8 + j;
                v16bf bb = *(const v16bf*)&Bs[p][(nt * 32 + lane) * 16];
                co[j] = __builtin_amdgcn_wmma_f32_16x16x32_bf16(false, a, false, bb,
                                                                (short)0, co[j], false, false);
            }
            __syncthreads();
        }
        #pragma unroll
        for (int j = 0; j < 8; ++j) {
            int n = nc2 * 256 + (nh * 8 + j) * 16 + (lane & 15);
            float b3v = b3e ? b3e[n] : 0.f;
            #pragma unroll
            for (int r = 0; r < 8; ++r) {
                int m = m16 * 16 + r + ((lane >> 4) << 3);
                float g = gateS[m];
                int tok = tokS[m];
                unsafeAtomicAdd(&y[(long)tok * DDIM + n], g * (co[j][r] + b3v));
            }
        }
    }
}

// ---------------------------------------------------------------------------
extern "C" void kernel_launch(void* const* d_in, const int* in_sizes, int n_in,
                              void* d_out, int out_size, void* d_ws, size_t ws_size,
                              hipStream_t stream) {
    const float* x   = (const float*)d_in[0];
    const float* rw  = (const float*)d_in[1];
    const float* rb  = (const float*)d_in[2];
    const float* W1  = (const float*)d_in[3];
    const float* b1  = (const float*)d_in[4];
    const float* W2  = (const float*)d_in[5];
    const float* b2  = (const float*)d_in[6];
    const float* W3  = (const float*)d_in[7];
    const float* b3  = (const float*)d_in[8];
    const float* sw1 = (const float*)d_in[9];
    const float* sb1 = (const float*)d_in[10];
    const float* sw2 = (const float*)d_in[11];
    const float* sb2 = (const float*)d_in[12];
    const float* sw3 = (const float*)d_in[13];
    const float* sb3 = (const float*)d_in[14];
    float* y = (float*)d_out;

    // workspace carve-up (<= ~36.2 MB)
    char* ws = (char*)d_ws;
    int*    cnt        = (int*)(ws + 0);
    int*    paddedCnt  = (int*)(ws + 256);
    int*    rowStart   = (int*)(ws + 512);
    int*    tlist      = (int*)(ws + 1024);                 // 32*2048 int
    float*  glist      = (float*)(ws + 1024 + 262144);      // 32*2048 f32
    __bf16* Abuf       = (__bf16*)(ws + 525312);            // routed A, <=30 MB
    __bf16* Ashared    = (__bf16*)(ws + 31982592);          // dense A, 4 MB

    moe_init<<<dim3((T_TOK * DDIM + 255) / 256), 256, 0, stream>>>(y, sb3, cnt);
    moe_router<<<dim3(T_TOK), 32, 0, stream>>>(x, rw, rb, cnt, tlist, glist);
    moe_scan<<<dim3(1), 32, 0, stream>>>(cnt, paddedCnt, rowStart);

    moe_pack_a<<<dim3(NEXP, T_TOK / TM), 256, 0, stream>>>(
        x, Abuf, rowStart, cnt, paddedCnt, 0, tlist, glist);
    moe_pack_a<<<dim3(1, T_TOK / TM), 256, 0, stream>>>(
        x, Ashared, nullptr, nullptr, nullptr, T_TOK, nullptr, nullptr);

    // routed experts: W1/W2 [E][D][I] (ld=I=512), W3 [E][I][D], b3 [E][D]
    moe_expert_gemm<IDIM><<<dim3(NEXP, T_TOK / TM), 256, 0, stream>>>(
        Abuf, rowStart, paddedCnt, 0, tlist, glist,
        W1, (long)DDIM * IDIM,
        W2, (long)DDIM * IDIM,
        b1, (long)IDIM, b2,
        W3, (long)IDIM * DDIM,
        b3, (long)DDIM, y);

    // shared expert as 2 pseudo-experts: sw1/sw2 [D][SI] column-sliced
    // (per-expert stride 512, ld=SI=1024); sw3 [SI][D] row-sliced; no b3.
    moe_expert_gemm<2 * IDIM><<<dim3(2, T_TOK / TM), 256, 0, stream>>>(
        Ashared, nullptr, nullptr, T_TOK, nullptr, nullptr,
        sw1, (long)IDIM,
        sw2, (long)IDIM,
        sb1, (long)IDIM, sb2,
        sw3, (long)IDIM * DDIM,
        nullptr, 0L, y);
}
